// KernelAttention_32590211842612
// MI455X (gfx1250) — matrix operand order repaired and checked by reading, not compiled, not run on hardware
//
#include <hip/hip_runtime.h>
#include <math.h>

typedef __attribute__((ext_vector_type(16))) __bf16 v16bf;
typedef __attribute__((ext_vector_type(8)))  float  v8f;
typedef __attribute__((ext_vector_type(4)))  __bf16 v4bf;

#define DIM 64
#define RES 128
#define WIN 16
#define WS  32
#define NPIX (RES*RES)          // 16384
#define KTOT (9*DIM)            // 576 = taps * channels
#define PRES (RES+2)            // 130 : 1-pixel zero halo
#define PPIX (PRES*PRES)        // 16900

// ---------------------------------------------------------------------------
// K0: convert x (f32, channel-innermost) into a zero-haloed bf16 image
//     xpad[b, (h+1)*130 + (w+1), c].  One thread per padded pixel (64 ch).
// ---------------------------------------------------------------------------
__global__ void pad_cvt_kernel(const float* __restrict__ x,
                               __bf16* __restrict__ xpad, int total) {
    int idx = blockIdx.x * blockDim.x + threadIdx.x;   // b*PPIX + pp
    if (idx >= total) return;
    int pp = idx % PPIX;
    int b  = idx / PPIX;
    int ph = pp / PRES, pw = pp % PRES;
    int h = ph - 1, w = pw - 1;
    v4bf* op = (v4bf*)(xpad + (size_t)idx * DIM);
    if ((unsigned)h < (unsigned)RES && (unsigned)w < (unsigned)RES) {
        const float4* ip =
            (const float4*)(x + (size_t)(b * NPIX + h * RES + w) * DIM);
        #pragma unroll
        for (int k = 0; k < 16; ++k) {
            float4 f = ip[k];
            v4bf o;
            o[0] = (__bf16)f.x; o[1] = (__bf16)f.y;
            o[2] = (__bf16)f.z; o[3] = (__bf16)f.w;
            op[k] = o;
        }
    } else {
        v4bf z = {};
        #pragma unroll
        for (int k = 0; k < 16; ++k) op[k] = z;
    }
}

// ---------------------------------------------------------------------------
// K1: pooled[b,wi,d] = mean over 32x32 window of depthwise-3x3(x) output
//     = (1/1024) * sum_{i,j} conv1_w[wi*64+d, i, j] * S[i-1][j-1]
//     where S[di][dj] = shifted partial sums of the window (zero padding).
// grid: B*WIN blocks x 256 threads (d = tid%64, row-quarter q = tid/64)
// ---------------------------------------------------------------------------
__global__ void pooled_kernel(const float* __restrict__ x,
                              const float* __restrict__ conv1_w,
                              float* __restrict__ pooled) {
    int bw = blockIdx.x;
    int wi = bw & 15;
    int b  = bw >> 4;
    int t  = threadIdx.x;
    int d  = t & 63;
    int q  = t >> 6;
    int h0 = (wi >> 2) * WS, w0 = (wi & 3) * WS;

    float S[9] = {0.f,0.f,0.f,0.f,0.f,0.f,0.f,0.f,0.f};
    for (int rr = 0; rr < 8; ++rr) {
        int r = q * 8 + rr;
        float rw0 = (r <= 30) ? 1.f : 0.f;   // di = -1 uses rows 0..30
        float rw2 = (r >= 1)  ? 1.f : 0.f;   // di = +1 uses rows 1..31
        const float* xrow = x + (((size_t)b * NPIX + (size_t)(h0 + r) * RES + w0) * DIM) + d;
        for (int cc = 0; cc < 32; ++cc) {
            float xv = xrow[(size_t)cc * DIM];
            float cw0 = (cc <= 30) ? 1.f : 0.f;
            float cw2 = (cc >= 1)  ? 1.f : 0.f;
            S[0] += rw0 * cw0 * xv; S[1] += rw0 * xv; S[2] += rw0 * cw2 * xv;
            S[3] += cw0 * xv;       S[4] += xv;       S[5] += cw2 * xv;
            S[6] += rw2 * cw0 * xv; S[7] += rw2 * xv; S[8] += rw2 * cw2 * xv;
        }
    }
    __shared__ float sm[256][9];
    for (int k = 0; k < 9; ++k) sm[t][k] = S[k];
    __syncthreads();
    if (q == 0) {
        float acc = 0.f;
        const float* kw = conv1_w + (size_t)(wi * DIM + d) * 9;
        for (int k = 0; k < 9; ++k) {
            float Tk = sm[d][k] + sm[d + 64][k] + sm[d + 128][k] + sm[d + 192][k];
            acc += kw[k] * Tk;
        }
        pooled[(b * WIN + wi) * DIM + d] = acc * (1.0f / 1024.0f);
    }
}

// ---------------------------------------------------------------------------
// K2: per-batch gate MLP: pooled -> grouped 1x1 -> GELU(exact) -> sigmoid
// grid: B blocks x 64 threads
// ---------------------------------------------------------------------------
__global__ void mlp_kernel(const float* __restrict__ pooled,
                           const float* __restrict__ dc_w, const float* __restrict__ dc_b,
                           const float* __restrict__ l1_w, const float* __restrict__ l1_b,
                           const float* __restrict__ l2_w, const float* __restrict__ l2_b,
                           float* __restrict__ sbuf) {
    int b = blockIdx.x;
    int t = threadIdx.x;           // 0..63
    __shared__ float v[WIN], hb[64];
    if (t < WIN) {
        float acc = dc_b[t];
        const float* pr = pooled + (b * WIN + t) * DIM;
        const float* wr = dc_w + t * DIM;
        for (int d = 0; d < DIM; ++d) acc += pr[d] * wr[d];
        v[t] = acc;
    }
    __syncthreads();
    {
        float acc = l1_b[t];
        const float* wr = l1_w + t * WIN;
        for (int wi = 0; wi < WIN; ++wi) acc += wr[wi] * v[wi];
        hb[t] = 0.5f * acc * (1.0f + erff(acc * 0.70710678118654752440f));
    }
    __syncthreads();
    if (t < WIN) {
        float acc = l2_b[t];
        const float* wr = l2_w + t * 64;
        for (int j = 0; j < 64; ++j) acc += wr[j] * hb[j];
        sbuf[b * WIN + t] = 1.0f / (1.0f + expf(-acc));
    }
}

// ---------------------------------------------------------------------------
// K3: effective dense 3x3 kernel, stored bf16, k-major with k = tap*64 + d
//     W_eff[b,o,k=t*64+d] =
//       sum_wi (fus_w[o,wi*64+d] + fus_w[o,1024+d]*gk_w[wi]) * s[b,wi] * base[wi,d,t]
//       + fus_w[o,1024+d]*gk_b
// grid: B*64*64 threads
// ---------------------------------------------------------------------------
__global__ void weff_kernel(const float* __restrict__ fus_w,
                            const float* __restrict__ gk_w,
                            const float* __restrict__ gk_b,
                            const float* __restrict__ conv1_w,
                            const float* __restrict__ sbuf,
                            __bf16* __restrict__ wef) {
    int idx = blockIdx.x * blockDim.x + threadIdx.x;
    if (idx >= 4 * DIM * DIM) return;
    int d = idx & 63;
    int o = (idx >> 6) & 63;
    int b = idx >> 12;
    float fo16 = fus_w[o * ((WIN + 1) * DIM) + WIN * DIM + d];
    float gb = gk_b[0];
    float acc[9];
    for (int ij = 0; ij < 9; ++ij) acc[ij] = fo16 * gb;
    for (int wi = 0; wi < WIN; ++wi) {
        float coef = (fus_w[o * ((WIN + 1) * DIM) + wi * DIM + d] + fo16 * gk_w[wi])
                     * sbuf[b * WIN + wi];
        const float* bp = conv1_w + (size_t)(wi * DIM + d) * 9;
        for (int ij = 0; ij < 9; ++ij) acc[ij] += coef * bp[ij];
    }
    __bf16* wr = wef + (size_t)(b * DIM + o) * KTOT;
    for (int ij = 0; ij < 9; ++ij) wr[ij * DIM + d] = (__bf16)acc[ij];
}

// ---------------------------------------------------------------------------
// K4: dense 3x3 conv via implicit GEMM on v_wmma_f32_16x16x32_bf16
//     D[16 o x 16 pix] += A[16 o x 32 k] * B[32 k x 16 pix],  K = 576
// One wave: 16 o-channels x 64 pixels (4 N-tiles share each A fragment).
// Haloed input => every load is one base pointer + constant imm offset.
// Jobs: b(4) * otile(4) * pixgroup(256 of 64 pixels) = 4096 waves.
// ---------------------------------------------------------------------------
__global__ void conv_wmma_kernel(const __bf16* __restrict__ xpad,
                                 const __bf16* __restrict__ wef,
                                 const float* __restrict__ fus_b,
                                 float* __restrict__ out) {
    const int lane = threadIdx.x & 31;
    const int wave = threadIdx.x >> 5;
    int job = blockIdx.x * (blockDim.x >> 5) + wave;   // 0..4095
    int pg  = job & 255;
    int ot  = (job >> 8) & 3;
    int b   = job >> 10;
    int h   = pg >> 1;
    int w0  = (pg & 1) << 6;                           // 0 or 64

    const int n  = lane & 15;                          // pixel col / A row id
    const int c  = lane >> 4;                          // lane half
    const int obase = ot * 16 + 8 * c;                 // D rows this lane holds

    // accumulators pre-loaded with fusion bias
    v8f acc[4];
    #pragma unroll
    for (int v = 0; v < 8; ++v) {
        float bv = fus_b[obase + v];
        acc[0][v] = bv; acc[1][v] = bv; acc[2][v] = bv; acc[3][v] = bv;
    }

    // per-lane base pointers; all loop offsets below are compile-time consts
    const __bf16* arow  = wef + (size_t)(b * DIM + ot * 16 + n) * KTOT + 8 * c;
    const __bf16* bbase = xpad +
        ((size_t)(b * PPIX + (h + 1) * PRES + (w0 + n + 1)) * DIM + 8 * c);

    #pragma unroll
    for (int s = 0; s < 18; ++s) {                     // K steps of 32
        const int t     = s >> 1;
        const int dbase = (s & 1) << 5;                // channel base 0 / 32
        const int dy    = t / 3 - 1, dx = t % 3 - 1;

        // ---- A fragment: lane holds k = 32s + {8c..8c+7} and {8c+16..8c+23}
        v16bf afrag;
        reinterpret_cast<uint4*>(&afrag)[0] =
            *reinterpret_cast<const uint4*>(arow + 32 * s);
        reinterpret_cast<uint4*>(&afrag)[1] =
            *reinterpret_cast<const uint4*>(arow + 32 * s + 16);

        // ---- 4 pixel tiles share this A fragment (constant-offset loads)
        #pragma unroll
        for (int u = 0; u < 4; ++u) {
            const __bf16* bp = bbase + ((dy * PRES + dx + u * 16) * DIM + dbase);
            v16bf bfrag;
            reinterpret_cast<uint4*>(&bfrag)[0] =
                *reinterpret_cast<const uint4*>(bp);
            reinterpret_cast<uint4*>(&bfrag)[1] =
                *reinterpret_cast<const uint4*>(bp + 16);
            acc[u] = __builtin_amdgcn_wmma_f32_16x16x32_bf16(
                         false, afrag, false, bfrag, (short)0, acc[u], false, false);
        }
    }

    // ---- store D: lane holds pixel column n, o-channels obase..obase+7
    float* obaseptr = out + ((size_t)(b * NPIX + h * RES + w0 + n) * DIM + obase);
    #pragma unroll
    for (int u = 0; u < 4; ++u) {
        float* op = obaseptr + u * 16 * DIM;
        reinterpret_cast<float4*>(op)[0] = reinterpret_cast<const float4*>(&acc[u])[0];
        reinterpret_cast<float4*>(op)[1] = reinterpret_cast<const float4*>(&acc[u])[1];
    }
}

// ---------------------------------------------------------------------------
extern "C" void kernel_launch(void* const* d_in, const int* in_sizes, int n_in,
                              void* d_out, int out_size, void* d_ws, size_t ws_size,
                              hipStream_t stream) {
    const float* x       = (const float*)d_in[0];
    const float* conv1_w = (const float*)d_in[1];
    const float* dc_w    = (const float*)d_in[2];
    const float* dc_b    = (const float*)d_in[3];
    const float* l1_w    = (const float*)d_in[4];
    const float* l1_b    = (const float*)d_in[5];
    const float* l2_w    = (const float*)d_in[6];
    const float* l2_b    = (const float*)d_in[7];
    const float* gk_w    = (const float*)d_in[8];
    const float* gk_b    = (const float*)d_in[9];
    const float* fus_w   = (const float*)d_in[10];
    const float* fus_b   = (const float*)d_in[11];
    float* out = (float*)d_out;

    char* ws = (char*)d_ws;
    __bf16* xpad   = (__bf16*)ws;                                   // 8,652,800 B
    __bf16* wef    = (__bf16*)(ws + 8652800);                       //   294,912 B
    float*  pooled = (float*)(ws + 8652800 + 294912);               //    16,384 B
    float*  sbuf   = (float*)(ws + 8652800 + 294912 + 16384);       //       256 B

    // 1) x -> zero-haloed bf16 image
    int npp = 4 * PPIX;                                             // 67,600
    pad_cvt_kernel<<<(npp + 255) / 256, 256, 0, stream>>>(x, xpad, npp);

    // 2) window pooled stats
    pooled_kernel<<<4 * WIN, 256, 0, stream>>>(x, conv1_w, pooled);

    // 3) gate MLP
    mlp_kernel<<<4, 64, 0, stream>>>(pooled, dc_w, dc_b, l1_w, l1_b, l2_w, l2_b, sbuf);

    // 4) effective dense 3x3 kernels (bf16, WMMA k-order)
    weff_kernel<<<(4 * DIM * DIM) / 256, 256, 0, stream>>>(fus_w, gk_w, gk_b,
                                                           conv1_w, sbuf, wef);

    // 5) fused dense conv + bias via WMMA (4096 wave jobs, 8 waves/block)
    conv_wmma_kernel<<<512, 256, 0, stream>>>(xpad, wef, fus_b, out);
}
